// GRUQNetwork_24154896073478
// MI455X (gfx1250) — compile-verified, matrix operand order checked by reading, not run on hardware
//
#include <hip/hip_runtime.h>
#include <hip/hip_bf16.h>

// ---------------- problem constants ----------------
#define T_    16384
#define OBS_  256
#define MLP_  1024
#define REC_  1024
#define ACT_  18

typedef __bf16 bf16_t;
typedef __attribute__((ext_vector_type(16))) __bf16 v16bf;
typedef __attribute__((ext_vector_type(8)))  __bf16 v8bf;
typedef __attribute__((ext_vector_type(8)))  float  v8f;
typedef __attribute__((ext_vector_type(4)))  unsigned int v4u;
typedef __attribute__((ext_vector_type(8)))  int v8i;
typedef __attribute__((ext_vector_type(4)))  int v4i;

// ---------------- f32 -> bf16 convert ----------------
__global__ void cvt_f32_bf16_kernel(const float* __restrict__ src,
                                    bf16_t* __restrict__ dst, int n) {
    int i = blockIdx.x * blockDim.x + threadIdx.x;
    int stride = gridDim.x * blockDim.x;
    for (; i < n; i += stride) dst[i] = (bf16_t)src[i];
}

// ---------------- TDM tile load: global [rows x K] bf16 tile -> LDS ----------------
// Loads a 128-row x 32-col bf16 tile whose top-left element is `gaddr`, row
// stride K elements, into LDS at byte offset `lds_off`. TDM pad feature inserts
// 16B after each 64B row -> LDS row stride 80B (= LDST elems), 16B-aligned and
// bank-conflict-free for the WMMA fragment ds_load_b128s.
#if defined(__clang_major__) && __clang_major__ >= 23
#define TDM_LOAD(g0, g1, g2, g3) \
    __builtin_amdgcn_tensor_load_to_lds((g0), (g1), (g2), (g3), (v8i)0, 0)
#else
#define TDM_LOAD(g0, g1, g2, g3) \
    __builtin_amdgcn_tensor_load_to_lds((g0), (g1), (g2), (g3), 0)
#endif

__device__ __forceinline__ void tdm_load_tile(unsigned lds_off,
                                              const bf16_t* gaddr,
                                              int rows_total, int K) {
    unsigned long long ga = (unsigned long long)gaddr;
    v4u g0;
    g0.x = 1u;                                        // count=1, user descriptor
    g0.y = lds_off;                                   // lds_addr (bytes)
    g0.z = (unsigned)(ga & 0xffffffffu);              // global_addr[31:0]
    g0.w = (unsigned)((ga >> 32) & 0x01ffffffu)       // global_addr[56:32]
         | (2u << 30);                                // type=2 ("image")
    v8i g1;
    // data_size=1 (2B) | pad_enable | pad_interval=3 (16 DW = 64B) | pad_amount=3 (4 DW = 16B)
    g1[0] = (int)((1u << 16) | (1u << 20) | (3u << 22) | (3u << 25));
    unsigned td0 = (unsigned)K;                       // tensor_dim0 (elems per row)
    unsigned td1 = (unsigned)rows_total;              // tensor_dim1 (rows)
    g1[1] = (int)((td0 & 0xffffu) << 16);             // abar_addr=0 | tensor_dim0 lo16
    g1[2] = (int)(((td0 >> 16) & 0xffffu) | ((td1 & 0xffffu) << 16));
    g1[3] = (int)(((td1 >> 16) & 0xffffu) | (32u << 16));   // tile_dim0 = 32
    g1[4] = (int)128u;                                // tile_dim1 = 128, tile_dim2 = 0
    g1[5] = (int)K;                                   // tensor_dim0_stride lo32
    g1[6] = 0;                                        // stride hi16 | dim1_stride lo16
    g1[7] = 0;
    v4i gz = (v4i)0;                                  // groups 2/3 unused (2D tile)
    TDM_LOAD(g0, g1, gz, gz);
}

// ---------------- WMMA GEMM: C[M,N] = act(A[M,K] * B[N,K]^T + bias) ----------------
// A, B bf16 row-major (B holds weight rows, i.e. we compute A * W^T).
// block = 256 threads (8 waves), block tile 128x128, K-step 32, TDM-fed
// double-buffered LDS. wave (wm 0..1, wn 0..3) -> 64x32 sub-tile = 4x2 WMMAs.
#define BM 128
#define BN 128
#define BK 32
#define LDST 40   // LDS row stride in bf16 elems (80B: 64B data + 16B TDM pad)

__device__ __forceinline__ v16bf load_frag(const bf16_t* base, int lane) {
    // 16-bit A-matrix 16x32 layout: lanes 0-15 rows, lanes 16-31 same rows K+8;
    // each lane holds K = {hk..hk+7, 16+hk..16+hk+7}, hk = (lane>=16)*8
    const bf16_t* p = base + (lane & 15) * LDST + ((lane >> 4) << 3);
    v8bf lo = *(const v8bf*)p;
    v8bf hi = *(const v8bf*)(p + 16);
    return __builtin_shufflevector(lo, hi, 0,1,2,3,4,5,6,7,8,9,10,11,12,13,14,15);
}

__device__ __forceinline__ float mish_act(float x) {
    float sp = log1pf(__expf(x));       // softplus
    return x * tanhf(sp);
}

__global__ __launch_bounds__(256) void wmma_gemm_bias_act_kernel(
    const bf16_t* __restrict__ A, const bf16_t* __restrict__ B,
    const float* __restrict__ bias, bf16_t* __restrict__ C,
    int M, int N, int K, int act)
{
    __shared__ bf16_t Abuf[2][BM * LDST];
    __shared__ bf16_t Bbuf[2][BM * LDST];

    const int tid  = threadIdx.x;
    const int lane = tid & 31;
    const int wave = tid >> 5;
    const int wm   = wave & 1;       // 0..1  -> 64 rows each
    const int wn   = wave >> 1;      // 0..3  -> 32 cols each
    const int m0   = blockIdx.x * BM;
    const int n0   = blockIdx.y * BN;

    v8f acc[4][2];
    #pragma unroll
    for (int i = 0; i < 4; ++i)
        #pragma unroll
        for (int j = 0; j < 2; ++j)
            #pragma unroll
            for (int e = 0; e < 8; ++e) acc[i][j][e] = 0.0f;

    const int ntiles = K / BK;
    const bf16_t* Abase = A + (size_t)m0 * K;
    const bf16_t* Bbase = B + (size_t)n0 * K;

    // prologue: wave 0 kicks the DMA for K-tile 0 into buffer 0
    if (wave == 0) {
        tdm_load_tile((unsigned)(unsigned long long)&Abuf[0][0], Abase, M, K);
        tdm_load_tile((unsigned)(unsigned long long)&Bbuf[0][0], Bbase, N, K);
    }

    int buf = 0;
    for (int kt = 0; kt < ntiles; ++kt) {
        if (wave == 0) __builtin_amdgcn_s_wait_tensorcnt(0);  // current tile landed
        __syncthreads();                                      // publish to all waves
        if (wave == 0 && kt + 1 < ntiles) {
            // overlap: DMA next K-tile into the other buffer while we compute
            int k0n = (kt + 1) * BK;
            tdm_load_tile((unsigned)(unsigned long long)&Abuf[buf ^ 1][0], Abase + k0n, M, K);
            tdm_load_tile((unsigned)(unsigned long long)&Bbuf[buf ^ 1][0], Bbase + k0n, N, K);
        }

        v16bf af[4], bfm[2];
        #pragma unroll
        for (int tm = 0; tm < 4; ++tm)
            af[tm] = load_frag(&Abuf[buf][(wm * 64 + tm * 16) * LDST], lane);
        #pragma unroll
        for (int tn = 0; tn < 2; ++tn)
            bfm[tn] = load_frag(&Bbuf[buf][(wn * 32 + tn * 16) * LDST], lane);

        #pragma unroll
        for (int tm = 0; tm < 4; ++tm)
            #pragma unroll
            for (int tn = 0; tn < 2; ++tn)
                acc[tm][tn] = __builtin_amdgcn_wmma_f32_16x16x32_bf16(
                    false, af[tm], false, bfm[tn], (short)0, acc[tm][tn], false, false);
        buf ^= 1;
    }

    // epilogue: C/D 16x16 layout -> n = lane&15, m = vgpr + 8*(lane>>4)
    const int lr = lane & 15;
    const int lh = lane >> 4;
    #pragma unroll
    for (int tm = 0; tm < 4; ++tm) {
        #pragma unroll
        for (int tn = 0; tn < 2; ++tn) {
            int gm = m0 + wm * 64 + tm * 16;
            int gn = n0 + wn * 32 + tn * 16 + lr;
            float bv = bias[gn];
            #pragma unroll
            for (int v = 0; v < 8; ++v) {
                int m = gm + v + lh * 8;
                float x = acc[tm][tn][v] + bv;
                if (act) x = mish_act(x);
                C[(size_t)m * N + gn] = (bf16_t)x;
            }
        }
    }
}

// ---------------- persistent GRU scan ----------------
// 128 workgroups x 192 threads. WG b owns h elements e in [b*8, b*8+8) and the
// matching 24 rows of w_hh (8 r-rows, 8 z-rows, 8 n-rows) resident in LDS (bf16).
// One grid barrier per step; h is double-buffered in global ws.
#define NWG_SCAN 128
#define EPW   8          // h elements per WG
#define SROWS 24         // 3 * EPW rows of w_hh
#define SPART 8          // threads per row
#define SSEG  (REC_ / SPART)   // 128 elems per partial dot

__global__ __launch_bounds__(192) void gru_scan_kernel(
    const float* __restrict__ w_hh, const float* __restrict__ b_n,
    const bf16_t* __restrict__ ig, const int* __restrict__ start,
    float* __restrict__ hbuf, bf16_t* __restrict__ states,
    unsigned* __restrict__ sync)
{
    __shared__ bf16_t wlds[SROWS][REC_];   // 48 KB
    __shared__ float  hl[REC_];            // 4 KB (reset-applied h for this step)
    __shared__ float  partial[SROWS][SPART];
    __shared__ float  bns[EPW];

    const int tid = threadIdx.x;
    const int b   = blockIdx.x;

    // preload this WG's w_hh shard as bf16
    for (int idx = tid; idx < SROWS * REC_; idx += 192) {
        int l = idx >> 10;            // local row 0..23
        int k = idx & (REC_ - 1);
        int gate = l / EPW;
        int eo   = l - gate * EPW;
        int grow = gate * REC_ + b * EPW + eo;
        wlds[l][k] = (bf16_t)w_hh[(size_t)grow * REC_ + k];
    }
    if (tid < EPW) bns[tid] = b_n[b * EPW + tid];
    __syncthreads();

    unsigned* cnt = sync;
    unsigned* gen = sync + 64;   // separate cache line
    unsigned gen_local = 0;
    const int l  = tid >> 3;     // row 0..23
    const int pp = tid & 7;      // partial 0..7

    for (int t = 0; t < T_; ++t) {
        const int par = t & 1;
        const float* hin = hbuf + par * REC_;
        const int rst = start[t];
        for (int i = tid; i < REC_; i += 192) hl[i] = rst ? 0.0f : hin[i];
        __syncthreads();

        // partial dot: row l, segment pp
        float a = 0.0f;
        const bf16_t* wr = &wlds[l][pp * SSEG];
        const float*  hs = &hl[pp * SSEG];
        #pragma unroll 8
        for (int k = 0; k < SSEG; ++k) a += (float)wr[k] * hs[k];
        partial[l][pp] = a;
        __syncthreads();

        if (tid < SROWS) {
            float s = 0.0f;
            #pragma unroll
            for (int j = 0; j < SPART; ++j) s += partial[tid][j];
            partial[tid][0] = s;            // hg for local row tid
        }
        __syncthreads();

        if (tid < EPW) {
            const int e = b * EPW + tid;
            float hgr = partial[tid][0];
            float hgz = partial[EPW + tid][0];
            float hgn = partial[2 * EPW + tid][0];
            const bf16_t* igt = ig + (size_t)t * (3 * REC_);
            float r = 1.0f / (1.0f + __expf(-((float)igt[e] + hgr)));
            float z = 1.0f / (1.0f + __expf(-((float)igt[REC_ + e] + hgz)));
            float n = tanhf((float)igt[2 * REC_ + e] + r * (hgn + bns[tid]));
            float hold = hl[e];
            float hnew = n + z * (hold - n);
            hbuf[(par ^ 1) * REC_ + e] = hnew;
            states[(size_t)t * REC_ + e] = (bf16_t)hnew;
            __threadfence();
        }
        __syncthreads();

        // grid barrier (sense via monotonically increasing generation)
        ++gen_local;
        if (tid == 0) {
            unsigned prev = __hip_atomic_fetch_add(cnt, 1u, __ATOMIC_ACQ_REL,
                                                   __HIP_MEMORY_SCOPE_AGENT);
            if (prev == NWG_SCAN - 1) {
                __hip_atomic_store(cnt, 0u, __ATOMIC_RELAXED, __HIP_MEMORY_SCOPE_AGENT);
                __hip_atomic_fetch_add(gen, 1u, __ATOMIC_RELEASE, __HIP_MEMORY_SCOPE_AGENT);
            } else {
                while (__hip_atomic_load(gen, __ATOMIC_ACQUIRE,
                                         __HIP_MEMORY_SCOPE_AGENT) < gen_local) {
                    __builtin_amdgcn_s_sleep(8);
                }
            }
        }
        __syncthreads();
    }
}

// ---------------- scan init: zero barrier state, seed h ----------------
__global__ void scan_init_kernel(const float* __restrict__ state,
                                 float* __restrict__ hbuf,
                                 unsigned* __restrict__ sync) {
    int i = threadIdx.x;
    if (i < 128) sync[i] = 0u;
    for (int k = i; k < REC_; k += blockDim.x) {
        hbuf[k]        = state[k];
        hbuf[REC_ + k] = 0.0f;
    }
}

// ---------------- head: out[t, a] = y2[t,:] . p3_w[a,:] + p3_b[a] ----------------
__global__ __launch_bounds__(32) void head_kernel(
    const bf16_t* __restrict__ y2, const float* __restrict__ w,
    const float* __restrict__ b, float* __restrict__ out)
{
    const int t = blockIdx.x;
    const int a = threadIdx.x;
    if (a >= ACT_) return;
    const bf16_t* yr = y2 + (size_t)t * MLP_;
    const float*  wr = w + (size_t)a * MLP_;
    float acc = b[a];
    #pragma unroll 8
    for (int k = 0; k < MLP_; ++k) acc += (float)yr[k] * wr[k];
    out[(size_t)t * ACT_ + a] = acc;
}

__global__ void copy_final_kernel(const float* __restrict__ h,
                                  float* __restrict__ out) {
    out[threadIdx.x] = h[threadIdx.x];
}

// ---------------- launch ----------------
extern "C" void kernel_launch(void* const* d_in, const int* in_sizes, int n_in,
                              void* d_out, int out_size, void* d_ws, size_t ws_size,
                              hipStream_t stream) {
    (void)in_sizes; (void)n_in; (void)out_size; (void)ws_size;
    const float* x     = (const float*)d_in[0];
    const float* state = (const float*)d_in[1];
    const int*   start = (const int*)d_in[2];
    /* d_in[3] = done (unused by reference) */
    const float* pre_w = (const float*)d_in[4];
    const float* pre_b = (const float*)d_in[5];
    const float* w_ih  = (const float*)d_in[6];
    const float* w_hh  = (const float*)d_in[7];
    const float* b_ih  = (const float*)d_in[8];
    const float* b_n   = (const float*)d_in[9];
    const float* p1_w  = (const float*)d_in[10];
    const float* p1_b  = (const float*)d_in[11];
    const float* p2_w  = (const float*)d_in[12];
    const float* p2_b  = (const float*)d_in[13];
    const float* p3_w  = (const float*)d_in[14];
    const float* p3_b  = (const float*)d_in[15];
    float* out = (float*)d_out;

    char* ws = (char*)d_ws;
    size_t off = 0;
    auto alloc = [&](size_t bytes) -> char* {
        char* p = ws + off;
        off += (bytes + 255) & ~(size_t)255;
        return p;
    };
    bf16_t* x_bf      = (bf16_t*)alloc((size_t)T_ * OBS_ * 2);        // 8 MB
    bf16_t* prew_bf   = (bf16_t*)alloc((size_t)MLP_ * OBS_ * 2);      // 0.5 MB
    bf16_t* wih_bf    = (bf16_t*)alloc((size_t)3 * REC_ * MLP_ * 2);  // 6 MB
    bf16_t* p1w_bf    = (bf16_t*)alloc((size_t)MLP_ * REC_ * 2);      // 2 MB
    bf16_t* p2w_bf    = (bf16_t*)alloc((size_t)MLP_ * MLP_ * 2);      // 2 MB
    bf16_t* feat_bf   = (bf16_t*)alloc((size_t)T_ * MLP_ * 2);        // 32 MB
    bf16_t* ig_bf     = (bf16_t*)alloc((size_t)T_ * 3 * REC_ * 2);    // 96 MB
    bf16_t* states_bf = (bf16_t*)alloc((size_t)T_ * REC_ * 2);        // 32 MB
    bf16_t* y1_bf     = (bf16_t*)alloc((size_t)T_ * MLP_ * 2);        // 32 MB
    bf16_t* y2_bf     = (bf16_t*)alloc((size_t)T_ * MLP_ * 2);        // 32 MB
    float*  hbuf      = (float*)alloc((size_t)2 * REC_ * 4);
    unsigned* sync    = (unsigned*)alloc(512);

    // 1) convert inputs/weights to bf16
    cvt_f32_bf16_kernel<<<2048, 256, 0, stream>>>(x, x_bf, T_ * OBS_);
    cvt_f32_bf16_kernel<<<512, 256, 0, stream>>>(pre_w, prew_bf, MLP_ * OBS_);
    cvt_f32_bf16_kernel<<<2048, 256, 0, stream>>>(w_ih, wih_bf, 3 * REC_ * MLP_);
    cvt_f32_bf16_kernel<<<1024, 256, 0, stream>>>(p1_w, p1w_bf, MLP_ * REC_);
    cvt_f32_bf16_kernel<<<1024, 256, 0, stream>>>(p2_w, p2w_bf, MLP_ * MLP_);

    // 2) init scan state (barrier counters + h seed)
    scan_init_kernel<<<1, 256, 0, stream>>>(state, hbuf, sync);

    // 3) feat = mish(x @ pre_w^T + pre_b)   [T, MLP]
    wmma_gemm_bias_act_kernel<<<dim3(T_ / BM, MLP_ / BN), 256, 0, stream>>>(
        x_bf, prew_bf, pre_b, feat_bf, T_, MLP_, OBS_, 1);

    // 4) igates = feat @ w_ih^T + b_ih      [T, 3*REC]
    wmma_gemm_bias_act_kernel<<<dim3(T_ / BM, (3 * REC_) / BN), 256, 0, stream>>>(
        feat_bf, wih_bf, b_ih, ig_bf, T_, 3 * REC_, MLP_, 0);

    // 5) sequential GRU scan (persistent, LDS-resident w_hh shards)
    gru_scan_kernel<<<NWG_SCAN, 192, 0, stream>>>(
        w_hh, b_n, ig_bf, start, hbuf, states_bf, sync);

    // 6) y1 = mish(states @ p1_w^T + p1_b)
    wmma_gemm_bias_act_kernel<<<dim3(T_ / BM, MLP_ / BN), 256, 0, stream>>>(
        states_bf, p1w_bf, p1_b, y1_bf, T_, MLP_, REC_, 1);

    // 7) y2 = mish(y1 @ p2_w^T + p2_b)
    wmma_gemm_bias_act_kernel<<<dim3(T_ / BM, MLP_ / BN), 256, 0, stream>>>(
        y1_bf, p2w_bf, p2_b, y2_bf, T_, MLP_, MLP_, 1);

    // 8) out = y2 @ p3_w^T + p3_b           [T, ACT]
    head_kernel<<<T_, 32, 0, stream>>>(y2_bf, p3_w, p3_b, out);

    // 9) final_state (after even T, final h is in buffer 0)
    copy_final_kernel<<<1, REC_, 0, stream>>>(hbuf + (T_ & 1) * REC_,
                                              out + (size_t)T_ * ACT_);
}
